// MambaDecoder_62964220559362
// MI455X (gfx1250) — matrix operand order repaired and checked by reading, not compiled, not run on hardware
//
#include <hip/hip_runtime.h>
#include <hip/hip_bf16.h>

// ---------------- problem constants ----------------
constexpr int NN = 8192;   // decoded reads (sequence length)
constexpr int MM = 8192;   // encoded reads
constexpr int DM = 128;    // d_model
constexpr int DI = 256;    // d_inner
constexpr int DH = 256;    // d_hidden
constexpr int DS = 16;     // d_state
constexpr int DR = 8;      // dt_rank
constexpr int CH = 64;     // scan chunks
constexpr int CL = 128;    // chunk length (CH*CL == NN)

typedef _Float16 v16h __attribute__((ext_vector_type(16)));
typedef _Float16 v8h  __attribute__((ext_vector_type(8)));
typedef float    v8f  __attribute__((ext_vector_type(8)));

// ---------------- workspace layout (bytes) ----------------
constexpr size_t OFF_XS    = 0;                                   // [NN,DI] f32 silu(conv(x))
constexpr size_t OFF_ZS    = OFF_XS    + (size_t)NN*DI*4;         // [NN,DI] f32 silu(z)
constexpr size_t OFF_DTRAW = OFF_ZS    + (size_t)NN*DI*4;         // [NN,DR] f32
constexpr size_t OFF_B     = OFF_DTRAW + (size_t)NN*DR*4;         // [NN,DS] f32
constexpr size_t OFF_C     = OFF_B     + (size_t)NN*DS*4;         // [NN,DS] f32
constexpr size_t OFF_DTV   = OFF_C     + (size_t)NN*DS*4;         // [NN,DI] f32 softplus dt
constexpr size_t OFF_HEND  = OFF_DTV   + (size_t)NN*DI*4;         // [CH,DI*DS] f32
constexpr size_t OFF_APROD = OFF_HEND  + (size_t)CH*DI*DS*4;      // [CH,DI*DS] f32
constexpr size_t OFF_HSTART= OFF_APROD + (size_t)CH*DI*DS*4;      // [CH,DI*DS] f32
constexpr size_t OFF_YDZ   = OFF_HSTART+ (size_t)CH*DI*DS*4;      // [NN,DI] f32
constexpr size_t OFF_DP    = OFF_YDZ   + (size_t)NN*DI*4;         // [NN,DM] f32 mamba out
constexpr size_t OFF_DPH   = OFF_DP    + (size_t)NN*DM*4;         // [NN,DM] f16
constexpr size_t OFF_ENCH  = OFF_DPH   + (size_t)NN*DM*2;         // [MM,DM] f16
constexpr size_t OFF_QWH   = OFF_ENCH  + (size_t)MM*DM*2;         // [DH,DM] f16
constexpr size_t OFF_KWH   = OFF_QWH   + (size_t)DH*DM*2;         // [DH,DM] f16
constexpr size_t OFF_QF    = OFF_KWH   + (size_t)DH*DM*2;         // [NN,DH] f32 queries
constexpr size_t OFF_QH    = OFF_QF    + (size_t)NN*DH*4;         // [NN,DH] f16
constexpr size_t OFF_KF    = OFF_QH    + (size_t)NN*DH*2;         // [MM,DH] f32 keys
constexpr size_t OFF_KH    = OFF_KF    + (size_t)MM*DH*4;         // [MM,DH] f16

__device__ __forceinline__ float sigmoidf_(float x) { return 1.0f / (1.0f + __expf(-x)); }

// ---------------- small fused scalar kernels ----------------

// xz = u @ in_proj_w.T ; x-> conv affine -> silu ; z-> silu
__global__ void k_inproj(const float* __restrict__ u, const float* __restrict__ W,
                         const float* __restrict__ cw, const float* __restrict__ cb,
                         float* __restrict__ xs, float* __restrict__ zs) {
  int idx = blockIdx.x * blockDim.x + threadIdx.x;
  if (idx >= NN * DI) return;
  int t = idx / DI, d = idx % DI;
  const float* ur = u + (size_t)t * DM;
  const float* wx = W + (size_t)d * DM;
  const float* wz = W + (size_t)(d + DI) * DM;
  float x = 0.f, z = 0.f;
#pragma unroll 8
  for (int k = 0; k < DM; ++k) { x += ur[k] * wx[k]; z += ur[k] * wz[k]; }
  float xc = x * cw[d] + cb[d];
  xs[idx] = xc * sigmoidf_(xc);
  zs[idx] = z * sigmoidf_(z);
}

// x_dbl = xs @ x_proj_w.T ; split into dtraw / B / C
__global__ void k_xproj(const float* __restrict__ xs, const float* __restrict__ W,
                        float* __restrict__ dtraw, float* __restrict__ Bm, float* __restrict__ Cm) {
  int idx = blockIdx.x * blockDim.x + threadIdx.x;
  if (idx >= NN * (DR + 2 * DS)) return;
  int t = idx / (DR + 2 * DS), j = idx % (DR + 2 * DS);
  const float* xr = xs + (size_t)t * DI;
  const float* wr = W + (size_t)j * DI;
  float a = 0.f;
#pragma unroll 8
  for (int k = 0; k < DI; ++k) a += xr[k] * wr[k];
  if (j < DR)            dtraw[(size_t)t * DR + j] = a;
  else if (j < DR + DS)  Bm[(size_t)t * DS + (j - DR)] = a;
  else                   Cm[(size_t)t * DS + (j - DR - DS)] = a;
}

// dt = softplus(dtraw @ dt_proj_w.T + dt_proj_b)
__global__ void k_dtproj(const float* __restrict__ dtraw, const float* __restrict__ W,
                         const float* __restrict__ bias, float* __restrict__ dtv) {
  int idx = blockIdx.x * blockDim.x + threadIdx.x;
  if (idx >= NN * DI) return;
  int t = idx / DI, d = idx % DI;
  const float* r = dtraw + (size_t)t * DR;
  const float* w = W + (size_t)d * DR;
  float a = bias[d];
#pragma unroll
  for (int k = 0; k < DR; ++k) a += r[k] * w[k];
  dtv[idx] = (a > 20.f) ? a : log1pf(__expf(a));
}

// Chunked linear-recurrence scan, phase A: per-chunk local scan from h0=0,
// tracking running product of deltaA. Thread = (chunk, d_inner channel).
__global__ void k_scanA(const float* __restrict__ dtv, const float* __restrict__ xs,
                        const float* __restrict__ Bm, const float* __restrict__ Alog,
                        float* __restrict__ hend, float* __restrict__ aprod) {
  int idx = blockIdx.x * blockDim.x + threadIdx.x;
  if (idx >= CH * DI) return;
  int d = idx % DI, c = idx / DI;
  float Ad[DS], h[DS], ap[DS];
#pragma unroll
  for (int n = 0; n < DS; ++n) { Ad[n] = -__expf(Alog[d * DS + n]); h[n] = 0.f; ap[n] = 1.f; }
  for (int s = 0; s < CL; ++s) {
    int t = c * CL + s;
    float dt = dtv[(size_t)t * DI + d];
    float dtx = dt * xs[(size_t)t * DI + d];
    const float* br = Bm + (size_t)t * DS;
#pragma unroll
    for (int n = 0; n < DS; ++n) {
      float da = __expf(dt * Ad[n]);
      h[n] = da * h[n] + dtx * br[n];
      ap[n] *= da;
    }
  }
  float* he = hend + (size_t)c * DI * DS + (size_t)d * DS;
  float* av = aprod + (size_t)c * DI * DS + (size_t)d * DS;
#pragma unroll
  for (int n = 0; n < DS; ++n) { he[n] = h[n]; av[n] = ap[n]; }
}

// phase B: sequentially combine chunk transfer functions -> per-chunk start states
__global__ void k_scanCombine(const float* __restrict__ hend, const float* __restrict__ aprod,
                              float* __restrict__ hstart) {
  int i = blockIdx.x * blockDim.x + threadIdx.x;
  if (i >= DI * DS) return;
  float h = 0.f;
  for (int c = 0; c < CH; ++c) {
    hstart[(size_t)c * DI * DS + i] = h;
    h = aprod[(size_t)c * DI * DS + i] * h + hend[(size_t)c * DI * DS + i];
  }
}

// phase C: replay chunks with correct start state; emit y fused with (y+D*x)*silu(z)
__global__ void k_scanC(const float* __restrict__ dtv, const float* __restrict__ xs,
                        const float* __restrict__ Bm, const float* __restrict__ Cm,
                        const float* __restrict__ Alog, const float* __restrict__ Dv,
                        const float* __restrict__ zs, const float* __restrict__ hstart,
                        float* __restrict__ ydz) {
  int idx = blockIdx.x * blockDim.x + threadIdx.x;
  if (idx >= CH * DI) return;
  int d = idx % DI, c = idx / DI;
  float Ad[DS], h[DS];
  const float* hs = hstart + (size_t)c * DI * DS + (size_t)d * DS;
#pragma unroll
  for (int n = 0; n < DS; ++n) { Ad[n] = -__expf(Alog[d * DS + n]); h[n] = hs[n]; }
  float Dd = Dv[d];
  for (int s = 0; s < CL; ++s) {
    int t = c * CL + s;
    float dt = dtv[(size_t)t * DI + d];
    float xv = xs[(size_t)t * DI + d];
    float dtx = dt * xv;
    const float* br = Bm + (size_t)t * DS;
    const float* cr = Cm + (size_t)t * DS;
    float y = 0.f;
#pragma unroll
    for (int n = 0; n < DS; ++n) {
      float da = __expf(dt * Ad[n]);
      h[n] = da * h[n] + dtx * br[n];
      y += h[n] * cr[n];
    }
    ydz[(size_t)t * DI + d] = (y + Dd * xv) * zs[(size_t)t * DI + d];
  }
}

// out = ydz @ out_proj_w.T
__global__ void k_outproj(const float* __restrict__ ydz, const float* __restrict__ W,
                          float* __restrict__ dp) {
  int idx = blockIdx.x * blockDim.x + threadIdx.x;
  if (idx >= NN * DM) return;
  int t = idx / DM, j = idx % DM;
  const float* yr = ydz + (size_t)t * DI;
  const float* wr = W + (size_t)j * DI;
  float a = 0.f;
#pragma unroll 8
  for (int k = 0; k < DI; ++k) a += yr[k] * wr[k];
  dp[idx] = a;
}

__global__ void k_cvt_f16(const float* __restrict__ src, _Float16* __restrict__ dst, int n) {
  int i = blockIdx.x * blockDim.x + threadIdx.x;
  if (i < n) dst[i] = (_Float16)src[i];
}

// ---------------- WMMA GEMM: C[M,N] = A[M,K] * B[N,K]^T (+bias) ----------------
// block: 256 threads = 8 waves (2 x 4). wave tile 64x64 (4x4 wmma 16x16x32).
// block tile: 128(M) x 256(N). K multiple of 32.

__device__ __forceinline__ v16h load_a_frag(const _Float16* __restrict__ A, int lda,
                                            int m0, int k0, int lane) {
  // ISA 16-bit A 16x32 layout: lanes 0-15 row m=lane hold K {0..7,16..23};
  // lanes 16-31 row m=lane-16 hold K {8..15,24..31}.
  int m = m0 + (lane & 15);
  int ks = (lane >> 4) << 3;
  const _Float16* p = A + (size_t)m * lda + k0 + ks;
  union { v16h v; v8h h[2]; } u;
  u.h[0] = *(const v8h*)(p);
  u.h[1] = *(const v8h*)(p + 16);
  return u.v;
}

__device__ __forceinline__ v16h load_b_frag(const _Float16* __restrict__ B, int ldb,
                                            int n0, int k0, int lane) {
  // B 32x16: lanes 0-15 column n=lane hold K {0..7,16..23};
  // lanes 16-31 column n=lane-16 hold K {8..15,24..31} (matches A).
  int n = n0 + (lane & 15);
  int ks = (lane >> 4) << 3;
  const _Float16* p = B + (size_t)n * ldb + k0 + ks;
  union { v16h v; v8h h[2]; } u;
  u.h[0] = *(const v8h*)(p);
  u.h[1] = *(const v8h*)(p + 16);
  return u.v;
}

__global__ __launch_bounds__(256) void k_gemm_wmma(const _Float16* __restrict__ A,
                                                   const _Float16* __restrict__ B,
                                                   const float* __restrict__ bias,
                                                   float* __restrict__ C,
                                                   int M, int N, int K) {
  const int lane = threadIdx.x & 31;
  const int w = threadIdx.x >> 5;
  const int wm = w & 1;        // 2 waves along M
  const int wn = w >> 1;       // 4 waves along N
  const int m0 = blockIdx.y * 128 + wm * 64;
  const int n0 = blockIdx.x * 256 + wn * 64;

  v8f acc[4][4] = {};
  for (int k0 = 0; k0 < K; k0 += 32) {
    v16h a[4], b[4];
#pragma unroll
    for (int i = 0; i < 4; ++i) a[i] = load_a_frag(A, K, m0 + 16 * i, k0, lane);
#pragma unroll
    for (int j = 0; j < 4; ++j) b[j] = load_b_frag(B, K, n0 + 16 * j, k0, lane);
    if (k0 + 32 < K) {
      // pull next K-tile of B toward the WGP (global_prefetch_b8)
      __builtin_prefetch(B + (size_t)(n0 + (lane & 15)) * K + k0 + 32, 0, 1);
    }
#pragma unroll
    for (int i = 0; i < 4; ++i)
#pragma unroll
      for (int j = 0; j < 4; ++j)
        acc[i][j] = __builtin_amdgcn_wmma_f32_16x16x32_f16(
            false, a[i], false, b[j], (short)0, acc[i][j], false, false);
  }

  // f32 C/D layout: vgpr r, lanes 0-15 -> (M=r, N=lane); lanes 16-31 -> (M=r+8, N=lane-16)
  const int cmB = (lane >> 4) * 8;
  const int cn = lane & 15;
#pragma unroll
  for (int i = 0; i < 4; ++i) {
#pragma unroll
    for (int j = 0; j < 4; ++j) {
      int col = n0 + 16 * j + cn;
      float bv = bias ? bias[col] : 0.f;
#pragma unroll
      for (int r = 0; r < 8; ++r) {
        int row = m0 + 16 * i + cmB + r;
        C[(size_t)row * N + col] = acc[i][j][r] + bv;
      }
    }
  }
}

// ---------------- in-place row softmax over [rows, cols] ----------------
__global__ void k_softmax(float* __restrict__ data, int cols) {
  __shared__ float red[256];
  const int tid = threadIdx.x;
  float* p = data + (size_t)blockIdx.x * cols;
  float m = -3.402823466e+38f;
  for (int c = tid; c < cols; c += 256) m = fmaxf(m, p[c]);
  red[tid] = m; __syncthreads();
  for (int s = 128; s > 0; s >>= 1) { if (tid < s) red[tid] = fmaxf(red[tid], red[tid + s]); __syncthreads(); }
  m = red[0]; __syncthreads();
  float sum = 0.f;
  for (int c = tid; c < cols; c += 256) { float e = __expf(p[c] - m); p[c] = e; sum += e; }
  red[tid] = sum; __syncthreads();
  for (int s = 128; s > 0; s >>= 1) { if (tid < s) red[tid] += red[tid + s]; __syncthreads(); }
  float inv = 1.f / red[0];
  for (int c = tid; c < cols; c += 256) p[c] *= inv;
}

// ---------------- launch ----------------
extern "C" void kernel_launch(void* const* d_in, const int* in_sizes, int n_in,
                              void* d_out, int out_size, void* d_ws, size_t ws_size,
                              hipStream_t stream) {
  const float* enc   = (const float*)d_in[0];
  const float* dec   = (const float*)d_in[1];
  const float* inW   = (const float*)d_in[2];
  const float* convW = (const float*)d_in[3];
  const float* convB = (const float*)d_in[4];
  const float* xW    = (const float*)d_in[5];
  const float* dtW   = (const float*)d_in[6];
  const float* dtB   = (const float*)d_in[7];
  const float* Alog  = (const float*)d_in[8];
  const float* Dv    = (const float*)d_in[9];
  const float* outW  = (const float*)d_in[10];
  const float* qW    = (const float*)d_in[11];
  const float* qB    = (const float*)d_in[12];
  const float* kW    = (const float*)d_in[13];
  const float* kB    = (const float*)d_in[14];

  char* ws = (char*)d_ws;
  float*    xs     = (float*)(ws + OFF_XS);
  float*    zs     = (float*)(ws + OFF_ZS);
  float*    dtraw  = (float*)(ws + OFF_DTRAW);
  float*    Bm     = (float*)(ws + OFF_B);
  float*    Cm     = (float*)(ws + OFF_C);
  float*    dtv    = (float*)(ws + OFF_DTV);
  float*    hend   = (float*)(ws + OFF_HEND);
  float*    aprod  = (float*)(ws + OFF_APROD);
  float*    hstart = (float*)(ws + OFF_HSTART);
  float*    ydz    = (float*)(ws + OFF_YDZ);
  float*    dp     = (float*)(ws + OFF_DP);
  _Float16* dpH    = (_Float16*)(ws + OFF_DPH);
  _Float16* encH   = (_Float16*)(ws + OFF_ENCH);
  _Float16* qWH    = (_Float16*)(ws + OFF_QWH);
  _Float16* kWH    = (_Float16*)(ws + OFF_KWH);
  float*    Qf     = (float*)(ws + OFF_QF);
  _Float16* Qh     = (_Float16*)(ws + OFF_QH);
  float*    Kf     = (float*)(ws + OFF_KF);
  _Float16* Kh     = (_Float16*)(ws + OFF_KH);
  float*    scores = (float*)d_out;

  const int T = 256;

  // ---- Mamba block ----
  k_inproj<<<(NN * DI + T - 1) / T, T, 0, stream>>>(dec, inW, convW, convB, xs, zs);
  k_xproj<<<(NN * (DR + 2 * DS) + T - 1) / T, T, 0, stream>>>(xs, xW, dtraw, Bm, Cm);
  k_dtproj<<<(NN * DI + T - 1) / T, T, 0, stream>>>(dtraw, dtW, dtB, dtv);
  k_scanA<<<(CH * DI + T - 1) / T, T, 0, stream>>>(dtv, xs, Bm, Alog, hend, aprod);
  k_scanCombine<<<(DI * DS + T - 1) / T, T, 0, stream>>>(hend, aprod, hstart);
  k_scanC<<<(CH * DI + T - 1) / T, T, 0, stream>>>(dtv, xs, Bm, Cm, Alog, Dv, zs, hstart, ydz);
  k_outproj<<<(NN * DM + T - 1) / T, T, 0, stream>>>(ydz, outW, dp);

  // ---- f16 conversions for WMMA operands ----
  k_cvt_f16<<<(NN * DM + T - 1) / T, T, 0, stream>>>(dp, dpH, NN * DM);
  k_cvt_f16<<<(MM * DM + T - 1) / T, T, 0, stream>>>(enc, encH, MM * DM);
  k_cvt_f16<<<(DH * DM + T - 1) / T, T, 0, stream>>>(qW, qWH, DH * DM);
  k_cvt_f16<<<(DH * DM + T - 1) / T, T, 0, stream>>>(kW, kWH, DH * DM);

  // ---- queries / keys via WMMA (K = 128) ----
  k_gemm_wmma<<<dim3(DH / 256, NN / 128), T, 0, stream>>>(dpH, qWH, qB, Qf, NN, DH, DM);
  k_gemm_wmma<<<dim3(DH / 256, MM / 128), T, 0, stream>>>(encH, kWH, kB, Kf, MM, DH, DM);
  k_cvt_f16<<<(NN * DH + T - 1) / T, T, 0, stream>>>(Qf, Qh, NN * DH);
  k_cvt_f16<<<(MM * DH + T - 1) / T, T, 0, stream>>>(Kf, Kh, MM * DH);

  // ---- scores = Q @ K^T via WMMA (dominant 34.4 GFLOP GEMM, K = 256) ----
  k_gemm_wmma<<<dim3(MM / 256, NN / 128), T, 0, stream>>>(Qh, Kh, nullptr, scores, NN, MM, DH);

  // ---- row softmax in place on d_out ----
  k_softmax<<<NN, T, 0, stream>>>(scores, MM);
}
